// MultiAttention_84988812853613
// MI455X (gfx1250) — compile-verified, matrix operand order checked by reading, not run on hardware
//
#include <hip/hip_runtime.h>
#include <hip/hip_bf16.h>

// MI455X / gfx1250, wave32. Mixed-precision MHA: bf16 WMMA (f32 accumulate),
// async global->LDS tile pipeline (ASYNCcnt), fused scores+mask+softmax.

typedef __bf16 bf16_t;
typedef bf16_t v16bf __attribute__((ext_vector_type(16)));
typedef float  v8f   __attribute__((ext_vector_type(8)));

#define D_MODEL 1024
#define N_HEADS 16
#define DK 64
#define BATCH 8
#define SEQ 1024
#define NEGV (-1e9f)

static __device__ inline v8f v8zero() {
  v8f z;
#pragma unroll
  for (int i = 0; i < 8; ++i) z[i] = 0.0f;
  return z;
}

static __device__ inline v8f wmma_bf16(v16bf a, v16bf b, v8f c) {
  // v_wmma_f32_16x16x32_bf16: D = A(16x32) * B(32x16) + C(16x16 f32)
  return __builtin_amdgcn_wmma_f32_16x16x32_bf16(false, a, false, b,
                                                 (short)0, c, false, false);
}

// CDNA5 async DMA: 16B per lane, global -> LDS, tracked by ASYNCcnt.
// lds_byte = LDS byte offset (low 32 bits of generic shared address).
static __device__ inline void async_b128(unsigned lds_byte, const void* g) {
  asm volatile("global_load_async_to_lds_b128 %0, %1, off"
               :: "v"(lds_byte), "v"(g) : "memory");
}
static __device__ inline void wait_async_le4() {
  asm volatile("s_wait_asynccnt 0x4" ::: "memory");
}
static __device__ inline void wait_async_0() {
  asm volatile("s_wait_asynccnt 0x0" ::: "memory");
}

// Load one 16(row)x32(K) bf16 operand fragment, stored row-major [row][k]
// with rowStride bf16 elements (LDS or global). Layout per ISA 7.12.2:
// lane&15 = row; kb = (lane>>4)*8; VGPR j holds K pair
// ((j>>2)<<4)+kb+((j&3)<<1)  =>  each half-fragment is 8 contiguous bf16.
static __device__ inline v16bf ld_frag(const bf16_t* p, int rowBase,
                                       int rowStride, int k0) {
  int lane = threadIdx.x & 31;
  const bf16_t* rp = p + (size_t)(rowBase + (lane & 15)) * rowStride + k0 +
                     ((lane >> 4) << 3);
  union { v16bf v; uint4 q[2]; } f;
  f.q[0] = *(const uint4*)(rp);
  f.q[1] = *(const uint4*)(rp + 16);
  return f.v;
}

// ---------------------------------------------------------------- converts

__global__ void k_cvt_bf16(const float* __restrict__ in,
                           bf16_t* __restrict__ out, int n) {
  int i = blockIdx.x * blockDim.x + threadIdx.x;
  int stride = gridDim.x * blockDim.x;
  for (; i < n; i += stride) out[i] = (bf16_t)in[i];
}

// W [K=1024][N=1024] f32  ->  Wt [N][K] bf16 (K-contiguous B operand)
__global__ __launch_bounds__(256) void k_cvt_w_t(const float* __restrict__ W,
                                                 bf16_t* __restrict__ Wt) {
  __shared__ float tile[32][33];
  int tx = threadIdx.x, ty = threadIdx.y;          // blockDim = (32, 8)
  int c0 = blockIdx.x * 32, r0 = blockIdx.y * 32;
#pragma unroll
  for (int i = 0; i < 4; ++i)
    tile[ty + i * 8][tx] = W[(size_t)(r0 + ty + i * 8) * D_MODEL + c0 + tx];
  __syncthreads();
#pragma unroll
  for (int i = 0; i < 4; ++i)
    Wt[(size_t)(c0 + ty + i * 8) * D_MODEL + r0 + tx] =
        (bf16_t)tile[tx][ty + i * 8];
}

// -------------------------------------------------------- generic WMMA GEMM
// C(MxN,f32) = A(bf16 [M][K]) * Bt(bf16 [N][K])^T
// Block tile 128x128, K-step 32, double-buffered async global->LDS pipeline.
// 8 waves as 4(M)x2(N); wave tile 32x64 = 2x4 WMMA tiles, 8 v_wmma/step.
// MODE 0: f32 out [M][N]
// MODE 1: bf16 out, Q/K layout  [B,H,L,DK]   (m->(b,l), n->(h,d))
// MODE 2: bf16 out, V^T layout  [B,H,DK,L]
#define TSTRIDE 40
#define TSLOTS (128 * TSTRIDE)

static __device__ inline void stage_tile(const bf16_t* __restrict__ A,
                                         const bf16_t* __restrict__ Bt,
                                         bf16_t* As, bf16_t* Bs, int m0,
                                         int n0, int K, int k0, int lrow,
                                         int lseg) {
  unsigned la = (unsigned)(size_t)(As + lrow * TSTRIDE + lseg);
  const bf16_t* ga = A + (size_t)(m0 + lrow) * K + k0 + lseg;
  async_b128(la, ga);
  async_b128(la + 16, ga + 8);
  unsigned lb = (unsigned)(size_t)(Bs + lrow * TSTRIDE + lseg);
  const bf16_t* gb = Bt + (size_t)(n0 + lrow) * K + k0 + lseg;
  async_b128(lb, gb);
  async_b128(lb + 16, gb + 8);
}

template <int MODE>
__global__ __launch_bounds__(256) void k_gemm_bf16(
    const bf16_t* __restrict__ A, const bf16_t* __restrict__ Bt,
    void* __restrict__ Cout, int M, int N, int K) {
  __shared__ bf16_t As[2 * TSLOTS];
  __shared__ bf16_t Bs[2 * TSLOTS];
  int t = threadIdx.x;
  int w = t >> 5;
  int m0 = blockIdx.y * 128;
  int n0 = blockIdx.x * 128;
  int wm = (w >> 1) * 32;
  int wn = (w & 1) * 64;

  v8f acc[2][4];
#pragma unroll
  for (int i = 0; i < 2; ++i)
#pragma unroll
    for (int j = 0; j < 4; ++j) acc[i][j] = v8zero();

  int lrow = t >> 1, lseg = (t & 1) * 16;

  // prologue: stage k0=0 into buffer 0
  stage_tile(A, Bt, As, Bs, m0, n0, K, 0, lrow, lseg);

  int cur = 0;
  for (int k0 = 0; k0 < K; k0 += 32) {
    int nxt = k0 + 32;
    if (nxt < K) {
      // buffer cur^1 was last read two barriers ago -> safe to overwrite
      stage_tile(A, Bt, As + (cur ^ 1) * TSLOTS, Bs + (cur ^ 1) * TSLOTS, m0,
                 n0, K, nxt, lrow, lseg);
      wait_async_le4();  // in-order: first 4 (current buffer) completed
    } else {
      wait_async_0();
    }
    __syncthreads();

    const bf16_t* Ab = As + cur * TSLOTS;
    const bf16_t* Bb = Bs + cur * TSLOTS;
    v16bf af[2], bfg[4];
#pragma unroll
    for (int mt = 0; mt < 2; ++mt)
      af[mt] = ld_frag(Ab, wm + mt * 16, TSTRIDE, 0);
#pragma unroll
    for (int nt = 0; nt < 4; ++nt)
      bfg[nt] = ld_frag(Bb, wn + nt * 16, TSTRIDE, 0);
#pragma unroll
    for (int mt = 0; mt < 2; ++mt)
#pragma unroll
      for (int nt = 0; nt < 4; ++nt)
        acc[mt][nt] = wmma_bf16(af[mt], bfg[nt], acc[mt][nt]);
    __syncthreads();
    cur ^= 1;
  }

  int lane = t & 31;
#pragma unroll
  for (int mt = 0; mt < 2; ++mt)
#pragma unroll
    for (int nt = 0; nt < 4; ++nt) {
      int nn = n0 + wn + nt * 16 + (lane & 15);
      int mb = m0 + wm + mt * 16 + ((lane >> 4) << 3);
#pragma unroll
      for (int j = 0; j < 8; ++j) {
        int mm = mb + j;
        float v = acc[mt][nt][j];
        if constexpr (MODE == 0) {
          ((float*)Cout)[(size_t)mm * N + nn] = v;
        } else {
          int b = mm >> 10, l = mm & 1023, h = nn >> 6, d = nn & 63;
          size_t idx = (MODE == 1)
              ? ((size_t)((b * N_HEADS + h) * SEQ + l) * DK + d)
              : ((size_t)((b * N_HEADS + h) * DK + d) * SEQ + l);
          ((bf16_t*)Cout)[idx] = (bf16_t)v;
        }
      }
    }
}

// ---------------------------------------- fused scores + mask + softmax
// One workgroup = (b, h, 16 query rows). Q panel in LDS; K fragments gathered
// from global via 16B loads (K_bh = 128KB/head, L2-resident; 192MB L2).
// Wave w covers key cols [w*128, w*128+128): 8 N-tiles x 2 K-steps = 16 wmma.
// Scores panel 16x1024 f32 in LDS. Mask is applied during the softmax max
// pass where each thread owns 64 consecutive columns of one row, so the mask
// bytes come in as four coalesced b128 loads (not per-element u8 gathers).
__global__ __launch_bounds__(256) void k_attn_softmax(
    const bf16_t* __restrict__ Qb, const bf16_t* __restrict__ Kb,
    const unsigned char* __restrict__ mask, float* __restrict__ attn) {
  __shared__ bf16_t Qs[16 * 80];
  __shared__ float Ss[16 * 1032];
  __shared__ float redA[16 * 16];
  __shared__ float redB[16 * 16];
  int t = threadIdx.x, w = t >> 5;
  int q0 = blockIdx.x * 16;
  int h = blockIdx.y, b = blockIdx.z;
  const bf16_t* Qbh = Qb + (size_t)(b * N_HEADS + h) * SEQ * DK;
  const bf16_t* Kbh = Kb + (size_t)(b * N_HEADS + h) * SEQ * DK;

  if (t < 128) {
    int r = t >> 3, s = (t & 7) * 8;
    *(uint4*)(Qs + r * 80 + s) = *(const uint4*)(Qbh + (q0 + r) * DK + s);
  }
  __syncthreads();

  v16bf a0 = ld_frag(Qs, 0, 80, 0);
  v16bf a1 = ld_frag(Qs, 0, 80, 32);
  v8f acc[8];
#pragma unroll
  for (int i = 0; i < 8; ++i) acc[i] = v8zero();
#pragma unroll
  for (int nt = 0; nt < 8; ++nt) {
    int nb = w * 128 + nt * 16;
    v16bf b0 = ld_frag(Kbh, nb, DK, 0);
    v16bf b1 = ld_frag(Kbh, nb, DK, 32);
    acc[nt] = wmma_bf16(a0, b0, acc[nt]);
    acc[nt] = wmma_bf16(a1, b1, acc[nt]);
  }

  int lane = t & 31;
  const float scale = 0.125f;  // 1/sqrt(64)
#pragma unroll
  for (int nt = 0; nt < 8; ++nt) {
    int col = w * 128 + nt * 16 + (lane & 15);
    int rb = (lane >> 4) << 3;
#pragma unroll
    for (int j = 0; j < 8; ++j)
      Ss[(rb + j) * 1032 + col] = acc[nt][j] * scale;
  }
  __syncthreads();

  // softmax: 16 threads per row, 64 consecutive cols each
  int row = t >> 4, seg = t & 15;
  float* sr = Ss + row * 1032 + seg * 64;

  // coalesced mask fetch: 64 bytes = 4 x b128 per thread
  union { uint4 q[4]; unsigned char c[64]; } mrow;
  {
    const uint4* mp = (const uint4*)(mask +
        ((size_t)b * SEQ + (q0 + row)) * SEQ + seg * 64);
#pragma unroll
    for (int i = 0; i < 4; ++i) mrow.q[i] = mp[i];
  }

  float lmax = -3.4e38f;
#pragma unroll 8
  for (int i = 0; i < 64; ++i) {
    float v = mrow.c[i] ? NEGV : sr[i];
    sr[i] = v;
    lmax = fmaxf(lmax, v);
  }
  redA[row * 16 + seg] = lmax;
  __syncthreads();
  float rmax = -3.4e38f;
#pragma unroll
  for (int i = 0; i < 16; ++i) rmax = fmaxf(rmax, redA[row * 16 + i]);
  float lsum = 0.0f;
#pragma unroll 8
  for (int i = 0; i < 64; ++i) {
    float e = __expf(sr[i] - rmax);
    sr[i] = e;
    lsum += e;
  }
  redB[row * 16 + seg] = lsum;
  __syncthreads();
  float rsum = 0.0f;
#pragma unroll
  for (int i = 0; i < 16; ++i) rsum += redB[row * 16 + i];
  float inv = 1.0f / rsum;
  float* dst = attn + ((size_t)(b * N_HEADS + h) * SEQ + (q0 + row)) * SEQ
               + seg * 64;
#pragma unroll 8
  for (int i = 0; i < 64; ++i) dst[i] = sr[i] * inv;
}

// ---------------------------------------------- context = attn @ V
// Per (b, h, 128 query rows). attn f32 -> bf16 on the fly into LDS.
// B operand = Vt[b,h][d][l] (K-contiguous). Wave w owns 16 rows x 64 cols
// = 4 WMMA tiles; 32 K-steps => 128 v_wmma per wave.
__global__ __launch_bounds__(256) void k_context(
    const float* __restrict__ attn, const bf16_t* __restrict__ Vt,
    bf16_t* __restrict__ ctx) {
  __shared__ bf16_t As[128 * TSTRIDE];
  __shared__ bf16_t Bs[64 * TSTRIDE];
  int t = threadIdx.x, w = t >> 5;
  int m0 = blockIdx.x * 128;
  int h = blockIdx.y, b = blockIdx.z;
  const float* Abh = attn + ((size_t)(b * N_HEADS + h) * SEQ + m0) * SEQ;
  const bf16_t* Vbh = Vt + (size_t)(b * N_HEADS + h) * DK * SEQ;

  v8f acc[4];
#pragma unroll
  for (int i = 0; i < 4; ++i) acc[i] = v8zero();

  int lrow = t >> 1, lseg = (t & 1) * 16;
  for (int k0 = 0; k0 < SEQ; k0 += 32) {
    const float4* ga = (const float4*)(Abh + (size_t)lrow * SEQ + k0 + lseg);
    bf16_t* sa = As + lrow * TSTRIDE + lseg;
#pragma unroll
    for (int i = 0; i < 4; ++i) {
      float4 x = ga[i];
      sa[i * 4 + 0] = (bf16_t)x.x;
      sa[i * 4 + 1] = (bf16_t)x.y;
      sa[i * 4 + 2] = (bf16_t)x.z;
      sa[i * 4 + 3] = (bf16_t)x.w;
    }
    if (t < 128) {
      const uint4* gb = (const uint4*)(Vbh + (size_t)lrow * SEQ + k0 + lseg);
      uint4* sb = (uint4*)(Bs + lrow * TSTRIDE + lseg);
      sb[0] = gb[0];
      sb[1] = gb[1];
    }
    __syncthreads();

    v16bf af = ld_frag(As, w * 16, TSTRIDE, 0);
#pragma unroll
    for (int nt = 0; nt < 4; ++nt) {
      v16bf bfg = ld_frag(Bs, nt * 16, TSTRIDE, 0);
      acc[nt] = wmma_bf16(af, bfg, acc[nt]);
    }
    __syncthreads();
  }

  int lane = t & 31;
#pragma unroll
  for (int nt = 0; nt < 4; ++nt) {
    int d = nt * 16 + (lane & 15);
    int rb = (lane >> 4) << 3;
#pragma unroll
    for (int j = 0; j < 8; ++j) {
      int l = m0 + w * 16 + rb + j;
      ctx[(size_t)(b * SEQ + l) * (N_HEADS * DK) + h * DK + d] =
          (bf16_t)acc[nt][j];
    }
  }
}

// ------------------------------------------- residual + LayerNorm per row
__global__ __launch_bounds__(256) void k_ln(
    const float* __restrict__ pre, const float* __restrict__ X,
    const float* __restrict__ gamma, const float* __restrict__ beta,
    float* __restrict__ out) {
  __shared__ float s1[256], s2[256];
  int r = blockIdx.x, t = threadIdx.x;
  const float* pr = pre + (size_t)r * D_MODEL;
  const float* xr = X + (size_t)r * D_MODEL;
  float a = 0.0f, q = 0.0f, vals[4];
#pragma unroll
  for (int i = 0; i < 4; ++i) {
    int c = t + i * 256;
    float v = pr[c] + xr[c];
    vals[i] = v;
    a += v;
    q += v * v;
  }
  s1[t] = a;
  s2[t] = q;
  __syncthreads();
  for (int off = 128; off > 0; off >>= 1) {
    if (t < off) { s1[t] += s1[t + off]; s2[t] += s2[t + off]; }
    __syncthreads();
  }
  float mu = s1[0] * (1.0f / D_MODEL);
  float var = s2[0] * (1.0f / D_MODEL) - mu * mu;
  float rs = rsqrtf(var + 1e-6f);
#pragma unroll
  for (int i = 0; i < 4; ++i) {
    int c = t + i * 256;
    out[(size_t)r * D_MODEL + c] = (vals[i] - mu) * rs * gamma[c] + beta[c];
  }
}

// ---------------------------------------------------------------- launcher

extern "C" void kernel_launch(void* const* d_in, const int* in_sizes, int n_in,
                              void* d_out, int out_size, void* d_ws,
                              size_t ws_size, hipStream_t stream) {
  (void)in_sizes; (void)n_in; (void)out_size; (void)ws_size;
  const float* X = (const float*)d_in[0];
  const unsigned char* mask = (const unsigned char*)d_in[1];  // np.bool_
  const float* W_Q = (const float*)d_in[2];
  const float* W_K = (const float*)d_in[3];
  const float* W_V = (const float*)d_in[4];
  const float* W_O = (const float*)d_in[5];
  const float* gamma = (const float*)d_in[6];
  const float* beta = (const float*)d_in[7];

  const size_t ML = (size_t)BATCH * SEQ;  // 8192 token rows
  float* out1 = (float*)d_out;            // [B,L,D_MODEL] f32
  float* attn = out1 + ML * D_MODEL;      // [B,H,L,L] f32

  // workspace carve (~126 MB)
  char* ws = (char*)d_ws;
  bf16_t* Xb  = (bf16_t*)ws; ws += ML * D_MODEL * sizeof(bf16_t);
  bf16_t* WQt = (bf16_t*)ws; ws += (size_t)D_MODEL * D_MODEL * sizeof(bf16_t);
  bf16_t* WKt = (bf16_t*)ws; ws += (size_t)D_MODEL * D_MODEL * sizeof(bf16_t);
  bf16_t* WVt = (bf16_t*)ws; ws += (size_t)D_MODEL * D_MODEL * sizeof(bf16_t);
  bf16_t* WOt = (bf16_t*)ws; ws += (size_t)D_MODEL * D_MODEL * sizeof(bf16_t);
  bf16_t* Qb  = (bf16_t*)ws; ws += ML * D_MODEL * sizeof(bf16_t);
  bf16_t* Kb  = (bf16_t*)ws; ws += ML * D_MODEL * sizeof(bf16_t);
  bf16_t* Vt  = (bf16_t*)ws; ws += ML * D_MODEL * sizeof(bf16_t);
  bf16_t* ctx = (bf16_t*)ws; ws += ML * D_MODEL * sizeof(bf16_t);
  float*  pre = (float*)ws;  ws += ML * D_MODEL * sizeof(float);

  // 1) convert X, transpose+convert weights
  k_cvt_bf16<<<4096, 256, 0, stream>>>(X, Xb, (int)(ML * D_MODEL));
  dim3 tb(32, 8), tg(32, 32);
  k_cvt_w_t<<<tg, tb, 0, stream>>>(W_Q, WQt);
  k_cvt_w_t<<<tg, tb, 0, stream>>>(W_K, WKt);
  k_cvt_w_t<<<tg, tb, 0, stream>>>(W_V, WVt);
  k_cvt_w_t<<<tg, tb, 0, stream>>>(W_O, WOt);

  // 2) Q/K/V projections (M=8192, N=1024, K=1024)
  dim3 gg(D_MODEL / 128, (unsigned)(ML / 128));
  k_gemm_bf16<1><<<gg, 256, 0, stream>>>(Xb, WQt, Qb, (int)ML, D_MODEL, D_MODEL);
  k_gemm_bf16<1><<<gg, 256, 0, stream>>>(Xb, WKt, Kb, (int)ML, D_MODEL, D_MODEL);
  k_gemm_bf16<2><<<gg, 256, 0, stream>>>(Xb, WVt, Vt, (int)ML, D_MODEL, D_MODEL);

  // 3) fused scores + mask + softmax -> attn (f32, to d_out)
  k_attn_softmax<<<dim3(SEQ / 16, N_HEADS, BATCH), 256, 0, stream>>>(
      Qb, Kb, mask, attn);

  // 4) context = attn @ V -> ctx bf16 [B,L,H*DK]
  k_context<<<dim3(SEQ / 128, N_HEADS, BATCH), 256, 0, stream>>>(attn, Vt, ctx);

  // 5) output projection -> pre (f32), then residual + LayerNorm -> out1
  k_gemm_bf16<0><<<gg, 256, 0, stream>>>(ctx, WOt, pre, (int)ML, D_MODEL, D_MODEL);
  k_ln<<<(unsigned)ML, 256, 0, stream>>>(pre, X, gamma, beta, out1);
}
